// MultiHeadedAttention_4054449128000
// MI455X (gfx1250) — compile-verified, matrix operand order checked by reading, not run on hardware
//
#include <hip/hip_runtime.h>
#include <hip/hip_bf16.h>

// ---------------------------------------------------------------------------
// MultiHeadedAttention (B=64, M=1, L=E=4096, H=64, DK=64) for MI455X/gfx1250.
// Bandwidth-bound: 268 MB fp32 weights -> ~11.5us floor @ 23.3 TB/s.
// fp32 coalesced loads -> v_cvt_pk_bf16_f32 -> LDS (double-buffered) ->
// v_wmma_f32_16x16x32_bf16. Small BN + K-split for WGP coverage.
// ---------------------------------------------------------------------------

typedef __bf16 v16bf __attribute__((ext_vector_type(16)));
typedef __bf16 v8bf  __attribute__((ext_vector_type(8)));
typedef float  v8f   __attribute__((ext_vector_type(8)));

#define Ldim 4096          // K of every big GEMM
#define Edim 4096          // N of every big GEMM
#define Mrows 64           // M of every big GEMM (batch)
#define BK 32              // K step (one bf16 WMMA)
#define BN 32              // N per block (2 waves x 16)
#define NTHREADS 64        // 2 waves (wave32)
#define LA_STRIDE 40       // 32 + 8 pad (bf16 units)
#define LB_STRIDE 40       // 32 + 8 pad (bf16 units)

// Out = X(64 x K-range) * W(K x N) [+ bias], fp32 memory, bf16 WMMA compute.
__device__ __forceinline__ void gemm_xw_body(
    const float* __restrict__ X,
    const float* __restrict__ W,
    const float* __restrict__ bias,   // nullptr -> no bias (partial output)
    float* __restrict__ Out,
    int nblk, int ks_begin, int ks_end)
{
    __shared__ __align__(16) __bf16 lA[2][Mrows * LA_STRIDE];
    __shared__ __align__(16) __bf16 lB[2][BK * LB_STRIDE];

    const int tid  = threadIdx.x;
    const int lane = tid & 31;
    const int wid  = tid >> 5;          // 0..1 : 16-wide N strip within block
    const int n0   = nblk * BN;

    // cooperative load mapping (64 threads)
    const int arow = tid;               // one A row per thread (32 floats)
    const int brow = tid >> 1;          // 0..31 (k within chunk)
    const int bn0  = (tid & 1) * 16;    // 0 or 16

    // WMMA 16-bit fragment K-chunk offsets (ISA 16-bit A 16x32 layout)
    const int k0a  = (lane >= 16) ? 8 : 0;
    const int k0b  = k0a + 16;
    const int frow = lane & 15;

    // stage one K-chunk into LDS buffer `buf`
    auto stage = [&](int buf, int ks) {
        // ---- X chunk: 64 rows x 32 k ----
        {
            const float4* src = (const float4*)(X + (size_t)arow * Ldim + ks);
            float4 f0 = src[0], f1 = src[1], f2 = src[2], f3 = src[3];
            float4 f4 = src[4], f5 = src[5], f6 = src[6], f7 = src[7];
            v8bf p0 = { (__bf16)f0.x,(__bf16)f0.y,(__bf16)f0.z,(__bf16)f0.w,
                        (__bf16)f1.x,(__bf16)f1.y,(__bf16)f1.z,(__bf16)f1.w };
            v8bf p1 = { (__bf16)f2.x,(__bf16)f2.y,(__bf16)f2.z,(__bf16)f2.w,
                        (__bf16)f3.x,(__bf16)f3.y,(__bf16)f3.z,(__bf16)f3.w };
            v8bf p2 = { (__bf16)f4.x,(__bf16)f4.y,(__bf16)f4.z,(__bf16)f4.w,
                        (__bf16)f5.x,(__bf16)f5.y,(__bf16)f5.z,(__bf16)f5.w };
            v8bf p3 = { (__bf16)f6.x,(__bf16)f6.y,(__bf16)f6.z,(__bf16)f6.w,
                        (__bf16)f7.x,(__bf16)f7.y,(__bf16)f7.z,(__bf16)f7.w };
            __bf16* dst = &lA[buf][arow * LA_STRIDE];
            *(v8bf*)&dst[0]  = p0;
            *(v8bf*)&dst[8]  = p1;
            *(v8bf*)&dst[16] = p2;
            *(v8bf*)&dst[24] = p3;
        }
        // ---- W chunk: 32 k x 32 n, row-major (k, n) ----
        {
            const float4* src = (const float4*)(W + (size_t)(ks + brow) * Edim + n0 + bn0);
            float4 f0 = src[0], f1 = src[1], f2 = src[2], f3 = src[3];
            v8bf p0 = { (__bf16)f0.x,(__bf16)f0.y,(__bf16)f0.z,(__bf16)f0.w,
                        (__bf16)f1.x,(__bf16)f1.y,(__bf16)f1.z,(__bf16)f1.w };
            v8bf p1 = { (__bf16)f2.x,(__bf16)f2.y,(__bf16)f2.z,(__bf16)f2.w,
                        (__bf16)f3.x,(__bf16)f3.y,(__bf16)f3.z,(__bf16)f3.w };
            __bf16* dst = &lB[buf][brow * LB_STRIDE + bn0];
            *(v8bf*)&dst[0] = p0;
            *(v8bf*)&dst[8] = p1;
        }
    };

    v8f acc[4];
#pragma unroll
    for (int m = 0; m < 4; ++m) acc[m] = (v8f){0.f,0.f,0.f,0.f,0.f,0.f,0.f,0.f};

    const int nsteps = (ks_end - ks_begin) / BK;
    stage(0, ks_begin);

    for (int i = 0; i < nsteps; ++i) {
        const int cur = i & 1;
        __syncthreads();   // buf[cur] stores visible; buf[cur^1] reads from i-1 done

        // software pipeline: issue next chunk's global loads + LDS stores now
        if (i + 1 < nsteps) stage(cur ^ 1, ks_begin + (i + 1) * BK);
        // prefetch W two chunks ahead into L2 (global_prefetch_b8)
        if (i + 2 < nsteps)
            __builtin_prefetch(W + (size_t)(ks_begin + (i + 2) * BK + brow) * Edim + n0 + bn0, 0, 1);

        // ---- B fragment: column n = wid*16 + frow, 16 K values ----
        v16bf bfrag;
        {
            const __bf16* lbp = &lB[cur][wid * 16 + frow];
#pragma unroll
            for (int j = 0; j < 8; ++j) {
                bfrag[j]     = lbp[(k0a + j) * LB_STRIDE];
                bfrag[8 + j] = lbp[(k0b + j) * LB_STRIDE];
            }
        }
        // ---- 4 M tiles, one bf16 WMMA each ----
#pragma unroll
        for (int m = 0; m < 4; ++m) {
            const __bf16* lap = &lA[cur][(m * 16 + frow) * LA_STRIDE];
            v8bf alo = *(const v8bf*)&lap[k0a];
            v8bf ahi = *(const v8bf*)&lap[k0b];
            v16bf afrag = __builtin_shufflevector(alo, ahi,
                0,1,2,3,4,5,6,7,8,9,10,11,12,13,14,15);
            acc[m] = __builtin_amdgcn_wmma_f32_16x16x32_bf16(
                false, afrag, false, bfrag, (short)0, acc[m], false, false);
        }
    }

    // ---- epilogue ----
    const int ncol  = n0 + wid * 16 + frow;
    const float bv  = bias ? bias[ncol] : 0.0f;
    const int mbase = (lane >> 4) * 8;   // C layout: lanes 16-31 hold M+8 rows
#pragma unroll
    for (int m = 0; m < 4; ++m) {
#pragma unroll
        for (int r = 0; r < 8; ++r) {
            int mm = m * 16 + mbase + r;
            Out[(size_t)mm * Edim + ncol] = acc[m][r] + bv;
        }
    }
}

// Stage 1: all three projections in one launch (grid.y selects q/k/v), full K.
__global__ __launch_bounds__(NTHREADS) void proj3_kernel(
    const float* q, const float* k, const float* v,
    const float* Wq, const float* Wk, const float* Wv,
    const float* bq, const float* bk, const float* bv,
    float* qh, float* kh, float* vh)
{
    const float *X, *W, *B; float* O;
    if (blockIdx.y == 0)      { X = q; W = Wq; B = bq; O = qh; }
    else if (blockIdx.y == 1) { X = k; W = Wk; B = bk; O = kh; }
    else                      { X = v; W = Wv; B = bv; O = vh; }
    gemm_xw_body(X, W, B, O, blockIdx.x, 0, Ldim);
}

// Stage 3a: out-GEMM, K split in half across grid.y for WGP coverage.
__global__ __launch_bounds__(NTHREADS) void out_gemm_partial_kernel(
    const float* attn, const float* Wo, float* p0, float* p1)
{
    float* O = blockIdx.y ? p1 : p0;
    const int klo = blockIdx.y * (Ldim / 2);
    gemm_xw_body(attn, Wo, nullptr, O, blockIdx.x, klo, klo + Ldim / 2);
}

// Stage 3b: out = p0 + p1 + bo  (bias broadcast along columns)
__global__ __launch_bounds__(256) void reduce_bias_kernel(
    const float* __restrict__ p0, const float* __restrict__ p1,
    const float* __restrict__ bo, float* __restrict__ out)
{
    int i = blockIdx.x * 256 + threadIdx.x;
    if (i < Mrows * Edim)
        out[i] = p0[i] + p1[i] + bo[i & (Edim - 1)];
}

// Stage 2a: dist[j,h] = softmax_h( cos(qh[j,h,:], kh[j,h,:]) )
__global__ __launch_bounds__(64) void cos_softmax_kernel(
    const float* __restrict__ qh, const float* __restrict__ kh,
    float* __restrict__ dist)
{
    const int j = blockIdx.x;
    const int h = threadIdx.x;
    const float* qp = qh + (size_t)j * Edim + h * 64;
    const float* kp = kh + (size_t)j * Edim + h * 64;
    float sqk = 0.f, sqq = 0.f, skk = 0.f;
#pragma unroll 4
    for (int d = 0; d < 64; ++d) {
        float a = qp[d], b = kp[d];
        sqk = fmaf(a, b, sqk);
        sqq = fmaf(a, a, sqq);
        skk = fmaf(b, b, skk);
    }
    float den = fmaxf(sqrtf(sqq) * sqrtf(skk), 1e-8f);
    float c = sqk / den;

    __shared__ float sh[64];
    __shared__ float sh2[64];
    sh[h] = c;
    __syncthreads();
    float mx = -1e30f;
    for (int i = 0; i < 64; ++i) mx = fmaxf(mx, sh[i]);
    float e = expf(c - mx);
    sh2[h] = e;
    __syncthreads();
    float s = 0.f;
    for (int i = 0; i < 64; ++i) s += sh2[i];
    dist[j * 64 + h] = e / s;
}

// Stage 2b: attn[b,j,d] = sum_h dist[j,h] * vh[b,h,d]   (one block per b)
__global__ __launch_bounds__(256) void attn_apply_kernel(
    const float* __restrict__ dist, const float* __restrict__ vh,
    float* __restrict__ attn)
{
    const int b = blockIdx.x;
    __shared__ float ld[64 * 64];
    __shared__ float lv[64 * 64];
    for (int i = threadIdx.x; i < 64 * 64; i += 256) {
        ld[i] = dist[i];
        lv[i] = vh[(size_t)b * Edim + i];
    }
    __syncthreads();
    for (int o = threadIdx.x; o < 64 * 64; o += 256) {
        const int j = o >> 6, d = o & 63;
        float s = 0.f;
#pragma unroll 8
        for (int h = 0; h < 64; ++h)
            s = fmaf(ld[j * 64 + h], lv[h * 64 + d], s);
        attn[(size_t)b * Edim + o] = s;
    }
}

extern "C" void kernel_launch(void* const* d_in, const int* in_sizes, int n_in,
                              void* d_out, int out_size, void* d_ws, size_t ws_size,
                              hipStream_t stream) {
    (void)in_sizes; (void)n_in; (void)out_size; (void)ws_size;

    const float* q  = (const float*)d_in[0];
    const float* k  = (const float*)d_in[1];
    const float* v  = (const float*)d_in[2];
    const float* Wq = (const float*)d_in[3];
    const float* bq = (const float*)d_in[4];
    const float* Wk = (const float*)d_in[5];
    const float* bk = (const float*)d_in[6];
    const float* Wv = (const float*)d_in[7];
    const float* bv = (const float*)d_in[8];
    const float* Wo = (const float*)d_in[9];
    const float* bo = (const float*)d_in[10];

    float* qh   = (float*)d_ws;                 // 64*4096
    float* kh   = qh + Mrows * Edim;
    float* vh   = kh + Mrows * Edim;
    float* attn = vh + Mrows * Edim;
    float* dist = attn + Mrows * Edim;          // 64*64
    float* p0   = dist + Mrows * Mrows;         // 64*4096 partial
    float* p1   = p0 + Mrows * Edim;            // 64*4096 partial

    dim3 blk(NTHREADS);
    proj3_kernel<<<dim3(Edim / BN, 3), blk, 0, stream>>>(
        q, k, v, Wq, Wk, Wv, bq, bk, bv, qh, kh, vh);

    cos_softmax_kernel<<<dim3(Mrows), dim3(64), 0, stream>>>(qh, kh, dist);
    attn_apply_kernel<<<dim3(Mrows), dim3(256), 0, stream>>>(dist, vh, attn);

    out_gemm_partial_kernel<<<dim3(Edim / BN, 2), blk, 0, stream>>>(attn, Wo, p0, p1);
    reduce_bias_kernel<<<dim3((Mrows * Edim + 255) / 256), dim3(256), 0, stream>>>(
        p0, p1, bo, (float*)d_out);
}
